// VGRUd_28346784153997
// MI455X (gfx1250) — matrix operand order repaired and checked by reading, not compiled, not run on hardware
//
#include <hip/hip_runtime.h>

// ---------------------------------------------------------------------------
// GRU sequence (B=32768, H=64, 25 steps) for gfx1250 (MI455X, wave32, WMMA).
//
// Strategy:
//  * prep kernel: fold weights into f16 column-major matrices:
//      M1 (192x256): first cell on [x|H]  -> [Wzr;Uzr | Wh;0 | 0;Uh]
//      M  ( 64x256): recurrent cell on h  -> [Wzr+Uzr | Wh | Uh]
//      Wdp( 64x80 ): logits, padded to 80 cols; bd padded with -1e30
//  * k_gru_first: one 16x192 @ 192x256 per wave (96 WMMA f16) -> h1 (f32, ws)
//  * k_gru_seq : wave owns 16 batch rows; h kept as f32 C-fragments in VGPRs;
//      per step: stage h->f16 LDS, rebuild A frags, 10 WMMA logits + softmax
//      (shfl_xor over 16-lane halves), 32 WMMA recurrent update.
// ---------------------------------------------------------------------------

typedef __attribute__((ext_vector_type(16))) _Float16 v16h;
typedef __attribute__((ext_vector_type(8)))  float    v8f;
typedef __attribute__((ext_vector_type(4)))  float    v4f;

union V16H { v16h v; v4f f4[2]; _Float16 h[16]; };

__device__ __forceinline__ v8f splat8(float s) {
  v8f v;
#pragma unroll
  for (int i = 0; i < 8; ++i) v[i] = s;
  return v;
}

__device__ __forceinline__ float fast_sigmoid(float x) {
  return 1.0f / (1.0f + __expf(-x));
}
__device__ __forceinline__ float fast_tanh(float x) {
  float ax = fabsf(x);
  float e  = __expf(2.0f * ax);
  float t  = 1.0f - 2.0f / (e + 1.0f);
  return copysignf(t, x);
}

__device__ __forceinline__ v8f wmma_f16(v16h a, v16h b, v8f c) {
  return __builtin_amdgcn_wmma_f32_16x16x32_f16(false, a, false, b,
                                                (short)0, c, false, false);
}

// B fragment (16-bit, 32x16) from column-major storage, column length 64.
// low lanes: K = kc*32 + {0..7,16..23} ; high lanes: K = kc*32 + {8..15,24..31} ; col = lane%16.
__device__ __forceinline__ v16h loadB_lds(const _Float16* base, int tileN,
                                          int kc, int lane) {
  int col  = tileN * 16 + (lane & 15);
  int half = lane >> 4;
  const _Float16* p = base + col * 64 + kc * 32 + half * 8;
  V16H b;
  b.f4[0] = *(const v4f*)p;
  b.f4[1] = *(const v4f*)(p + 16);
  return b.v;
}
// Same, column length 192 (first-cell matrix, read from global/L2).
__device__ __forceinline__ v16h loadB_g192(const _Float16* base, int tileN,
                                           int kc, int lane) {
  int col  = tileN * 16 + (lane & 15);
  int half = lane >> 4;
  const _Float16* p = base + col * 192 + kc * 32 + half * 8;
  V16H b;
  b.f4[0] = *(const v4f*)p;
  b.f4[1] = *(const v4f*)(p + 16);
  return b.v;
}

// ---------------------------------------------------------------------------
// Weight preparation: build f16 col-major matrices + fused biases in ws.
// Gate order in Wk/Uk columns: [z:0..63 | r:64..127 | h:128..191].
// Combined N layout (256): [z | r | xh | hh].
// ---------------------------------------------------------------------------
__global__ void k_prep(const float* __restrict__ Wk, const float* __restrict__ Uk,
                       const float* __restrict__ b,  const float* __restrict__ Wd,
                       const float* __restrict__ bd,
                       _Float16* __restrict__ M1, _Float16* __restrict__ M,
                       _Float16* __restrict__ Wdp, float* __restrict__ bias) {
  int tid = threadIdx.x;
  // M1: 256 cols x 192 rows, col-major M1[n*192 + k]
  for (int idx = tid; idx < 256 * 192; idx += blockDim.x) {
    int n = idx / 192, k = idx % 192;
    float v = 0.0f;
    if (n < 128)       v = (k < 128) ? Wk[k * 192 + n] : Uk[(k - 128) * 192 + n];
    else if (n < 192){ if (k < 128)  v = Wk[k * 192 + n]; }
    else             { if (k >= 128) v = Uk[(k - 128) * 192 + (n - 64)]; }
    M1[idx] = (_Float16)v;
  }
  // M: 256 cols x 64 rows, col-major M[n*64 + k]   (Wk64 = Wk[:64])
  for (int idx = tid; idx < 256 * 64; idx += blockDim.x) {
    int n = idx / 64, k = idx % 64;
    float v;
    if (n < 128)      v = Wk[k * 192 + n] + Uk[k * 192 + n];
    else if (n < 192) v = Wk[k * 192 + n];
    else              v = Uk[k * 192 + (n - 64)];
    M[idx] = (_Float16)v;
  }
  // Wdp: 80 cols x 64 rows, col-major; pad cols 78,79 with 0
  for (int idx = tid; idx < 80 * 64; idx += blockDim.x) {
    int n = idx / 64, k = idx % 64;
    Wdp[idx] = (_Float16)((n < 78) ? Wd[k * 78 + n] : 0.0f);
  }
  // biases: [0..127]=b0zr+b1zr, [128..191]=b0h, [192..255]=b1h, [256..335]=bd pad
  for (int n = tid; n < 128; n += blockDim.x) bias[n] = b[n] + b[192 + n];
  for (int n = tid; n < 64;  n += blockDim.x) bias[128 + n] = b[128 + n];
  for (int n = tid; n < 64;  n += blockDim.x) bias[192 + n] = b[320 + n];
  for (int n = tid; n < 80;  n += blockDim.x)
    bias[256 + n] = (n < 78) ? bd[n] : -1e30f;
}

// ---------------------------------------------------------------------------
// First GRU cell: h1 = GRU(x[128], H[64]).  Wave = 16 batch rows.
// [x|H] (16x192) @ M1 (192x256), B frags streamed from L2 (96KB, one-shot).
// ---------------------------------------------------------------------------
__global__ __launch_bounds__(128) void k_gru_first(
    const float* __restrict__ x, const float* __restrict__ H,
    const _Float16* __restrict__ M1, const float* __restrict__ bias,
    float* __restrict__ h1) {
  int tid  = threadIdx.x;
  int wave = tid >> 5, lane = tid & 31;
  int half = lane >> 4, m = lane & 15, n = m;
  int rowbase = blockIdx.x * 64 + wave * 16;
  int row = rowbase + m;

  // A fragments for [x|H] (K=192): per lane two 8-float runs per kc, f32->f16
  V16H a[6];
#pragma unroll
  for (int kc = 0; kc < 6; ++kc) {
    int c0 = kc * 32 + half * 8;
    int c1 = c0 + 16;
    const float* s0 = (c0 < 128) ? (x + row * 128 + c0) : (H + row * 64 + (c0 - 128));
    const float* s1 = (c1 < 128) ? (x + row * 128 + c1) : (H + row * 64 + (c1 - 128));
#pragma unroll
    for (int i = 0; i < 8; ++i) {
      a[kc].h[i]     = (_Float16)s0[i];
      a[kc].h[8 + i] = (_Float16)s1[i];
    }
  }

#pragma unroll
  for (int t = 0; t < 4; ++t) {
    v8f gz = splat8(bias[t * 16 + n]);
    v8f gr = splat8(bias[64 + t * 16 + n]);
    v8f gx = splat8(bias[128 + t * 16 + n]);
    v8f gh = splat8(bias[192 + t * 16 + n]);
#pragma unroll
    for (int kc = 0; kc < 6; ++kc) {
      gz = wmma_f16(a[kc].v, loadB_g192(M1, t,      kc, lane), gz);
      gr = wmma_f16(a[kc].v, loadB_g192(M1, 4 + t,  kc, lane), gr);
      gx = wmma_f16(a[kc].v, loadB_g192(M1, 8 + t,  kc, lane), gx);
      gh = wmma_f16(a[kc].v, loadB_g192(M1, 12 + t, kc, lane), gh);
    }
#pragma unroll
    for (int j = 0; j < 8; ++j) {
      int rr_row = rowbase + j + half * 8;
      float hp   = H[rr_row * 64 + t * 16 + n];
      float z    = fast_sigmoid(gz[j]);
      float rr   = fast_sigmoid(gr[j]);
      float cand = fast_tanh(gx[j] + rr * gh[j]);
      h1[rr_row * 64 + t * 16 + n] = z * hp + (1.0f - z) * cand;
    }
  }
}

// ---------------------------------------------------------------------------
// Recurrent steps + per-step logits/softmax.  Wave = 16 batch rows, 25 steps.
// ---------------------------------------------------------------------------
__global__ __launch_bounds__(256) void k_gru_seq(
    const _Float16* __restrict__ Mg, const _Float16* __restrict__ Wdg,
    const float* __restrict__ bias, const float* __restrict__ h1,
    float* __restrict__ out) {
  __shared__ __align__(16) _Float16 ldsM[256 * 64];   // 32 KB
  __shared__ __align__(16) _Float16 ldsW[80 * 64];    // 10 KB
  __shared__ __align__(16) _Float16 ldsH[8 * 16 * 64]; // 16 KB staging

  int tid = threadIdx.x;
  {  // cooperative weight load (uint4 = 16B)
    const uint4* s1 = (const uint4*)Mg;
    uint4*       d1 = (uint4*)ldsM;
    for (int i = tid; i < 256 * 64 / 8; i += 256) d1[i] = s1[i];
    const uint4* s2 = (const uint4*)Wdg;
    uint4*       d2 = (uint4*)ldsW;
    for (int i = tid; i < 80 * 64 / 8; i += 256) d2[i] = s2[i];
  }
  __syncthreads();

  int wave = tid >> 5, lane = tid & 31;
  int half = lane >> 4, m = lane & 15, n = m;
  int rowbase = blockIdx.x * 128 + wave * 16;

  float bzv[4], brv[4], bxv[4], bhv[4], bdv[5];
#pragma unroll
  for (int t = 0; t < 4; ++t) {
    bzv[t] = bias[t * 16 + n];
    brv[t] = bias[64 + t * 16 + n];
    bxv[t] = bias[128 + t * 16 + n];
    bhv[t] = bias[192 + t * 16 + n];
  }
#pragma unroll
  for (int lt = 0; lt < 5; ++lt) bdv[lt] = bias[256 + lt * 16 + n];

  // h state in f32 C-fragments (lane owns column t*16+n, VGPR j -> row j+8*half)
  v8f h[4];
#pragma unroll
  for (int t = 0; t < 4; ++t)
#pragma unroll
    for (int j = 0; j < 8; ++j)
      h[t][j] = h1[(rowbase + j + half * 8) * 64 + t * 16 + n];

  _Float16*       st = ldsH + wave * (16 * 64);
  const _Float16* rp = st + m * 64 + half * 8;

#pragma unroll 1
  for (int s = 0; s < 25; ++s) {
    // stage h (f32 -> f16, row-major per wave strip)
#pragma unroll
    for (int t = 0; t < 4; ++t)
#pragma unroll
      for (int j = 0; j < 8; ++j)
        st[(j + half * 8) * 64 + t * 16 + n] = (_Float16)h[t][j];

    // rebuild A fragments (K=64 -> 2 chunks of 32)
    V16H a0, a1;
    a0.f4[0] = *(const v4f*)(rp);
    a0.f4[1] = *(const v4f*)(rp + 16);
    a1.f4[0] = *(const v4f*)(rp + 32);
    a1.f4[1] = *(const v4f*)(rp + 48);

    // ---- logits (64 -> 80 padded) + softmax ----
    v8f L[5];
#pragma unroll
    for (int lt = 0; lt < 5; ++lt) {
      L[lt] = splat8(bdv[lt]);
      L[lt] = wmma_f16(a0.v, loadB_lds(ldsW, lt, 0, lane), L[lt]);
      L[lt] = wmma_f16(a1.v, loadB_lds(ldsW, lt, 1, lane), L[lt]);
    }
#pragma unroll
    for (int j = 0; j < 8; ++j) {
      float mx = L[0][j];
#pragma unroll
      for (int lt = 1; lt < 5; ++lt) mx = fmaxf(mx, L[lt][j]);
#pragma unroll
      for (int d = 1; d < 16; d <<= 1) mx = fmaxf(mx, __shfl_xor(mx, d, 16));
      float e[5];
      float sum = 0.0f;
#pragma unroll
      for (int lt = 0; lt < 5; ++lt) {
        e[lt] = __expf(L[lt][j] - mx);
        sum += e[lt];
      }
#pragma unroll
      for (int d = 1; d < 16; d <<= 1) sum += __shfl_xor(sum, d, 16);
      float inv = 1.0f / sum;
      float* orow = out + ((long)(rowbase + j + half * 8) * 25 + s) * 78;
#pragma unroll
      for (int lt = 0; lt < 5; ++lt) {
        int col = lt * 16 + n;
        if (col < 78) orow[col] = e[lt] * inv;
      }
    }

    // ---- recurrent update: G = h @ M (64x256), gate math, new h ----
    if (s < 24) {
#pragma unroll
      for (int t = 0; t < 4; ++t) {
        v8f gz = splat8(bzv[t]);
        v8f gr = splat8(brv[t]);
        v8f gx = splat8(bxv[t]);
        v8f gh = splat8(bhv[t]);
        gz = wmma_f16(a0.v, loadB_lds(ldsM, t, 0, lane), gz);
        gz = wmma_f16(a1.v, loadB_lds(ldsM, t, 1, lane), gz);
        gr = wmma_f16(a0.v, loadB_lds(ldsM, 4 + t, 0, lane), gr);
        gr = wmma_f16(a1.v, loadB_lds(ldsM, 4 + t, 1, lane), gr);
        gx = wmma_f16(a0.v, loadB_lds(ldsM, 8 + t, 0, lane), gx);
        gx = wmma_f16(a1.v, loadB_lds(ldsM, 8 + t, 1, lane), gx);
        gh = wmma_f16(a0.v, loadB_lds(ldsM, 12 + t, 0, lane), gh);
        gh = wmma_f16(a1.v, loadB_lds(ldsM, 12 + t, 1, lane), gh);
#pragma unroll
        for (int j = 0; j < 8; ++j) {
          float z    = fast_sigmoid(gz[j]);
          float rr   = fast_sigmoid(gr[j]);
          float cand = fast_tanh(gx[j] + rr * gh[j]);
          h[t][j]    = z * h[t][j] + (1.0f - z) * cand;
        }
      }
    }
  }
}

// ---------------------------------------------------------------------------
extern "C" void kernel_launch(void* const* d_in, const int* in_sizes, int n_in,
                              void* d_out, int out_size, void* d_ws,
                              size_t ws_size, hipStream_t stream) {
  const float* x  = (const float*)d_in[0];
  const float* H  = (const float*)d_in[1];
  const float* Wk = (const float*)d_in[2];
  const float* Uk = (const float*)d_in[3];
  const float* b  = (const float*)d_in[4];
  const float* Wd = (const float*)d_in[5];
  const float* bd = (const float*)d_in[6];
  float* out = (float*)d_out;

  int B = in_sizes[0] / 128;  // 32768

  char* ws = (char*)d_ws;
  _Float16* M1  = (_Float16*)(ws + 0);        // 256*192 f16 = 96 KB
  _Float16* M   = (_Float16*)(ws + 98304);    // 256*64  f16 = 32 KB
  _Float16* Wdp = (_Float16*)(ws + 131072);   // 80*64   f16 = 10 KB
  float*    bias= (float*)   (ws + 141312);   // 336 f32
  float*    h1  = (float*)   (ws + 147456);   // B*64 f32 = 8 MB

  k_prep<<<1, 256, 0, stream>>>(Wk, Uk, b, Wd, bd, M1, M, Wdp, bias);
  k_gru_first<<<B / 64, 128, 0, stream>>>(x, H, M1, bias, h1);
  k_gru_seq<<<B / 128, 256, 0, stream>>>(M, Wdp, bias, h1, out);
}
